// MultiHeadAttention_42906723287546
// MI455X (gfx1250) — matrix operand order repaired and checked by reading, not compiled, not run on hardware
//
#include <hip/hip_runtime.h>
#include <hip/hip_bf16.h>
#include <stddef.h>

// Problem constants (match reference)
#define BB 2
#define SS 2048
#define DD 1024
#define HH 16
#define DK 64

typedef __attribute__((ext_vector_type(16))) __bf16 v16bf;
typedef __attribute__((ext_vector_type(2)))  __bf16 v2bf;
typedef __attribute__((ext_vector_type(8)))  float  v8f;

union BF16x16 { v16bf v; unsigned int w[8]; unsigned short u[16]; };

// Pack two f32 -> two bf16 in one DWORD. __bf16 casts let the backend pick
// the native convert (v_cvt_pk_bf16_f32 on gfx1250).
__device__ __forceinline__ unsigned pk2bf(float a, float b) {
    v2bf p;
    p.x = (__bf16)a;
    p.y = (__bf16)b;
    unsigned r;
    __builtin_memcpy(&r, &p, 4);
    return r;
}

// Async copy of 16 bytes global -> LDS (per active lane). ASYNCcnt-tracked.
__device__ __forceinline__ void async_cp16(const float* gaddr, void* laddr) {
    unsigned lds = (unsigned)(size_t)laddr;              // low 32 bits = LDS offset
    unsigned long long g = (unsigned long long)(size_t)gaddr;
    asm volatile("global_load_async_to_lds_b128 %0, %1, off"
                 :: "v"(lds), "v"(g) : "memory");
}

// ---------------------------------------------------------------------------
// GEMM: Y[M,N] = X[M,K] @ W[K,N] + bias[N]   (bf16 WMMA, f32 accumulate)
// Block: 256 threads = 8 waves. Tile 128(M) x 64(N), K-chunk 32.
// Double-buffered async global->LDS staging of f32 tiles; f32->bf16 pair
// converts at fragment build. Wave (wr,wc) owns 32x32 = 2x2 WMMA tiles.
// ---------------------------------------------------------------------------
#define GTM 128
#define GTN 64
#define GTK 32

__global__ __launch_bounds__(256) void gemm_bias_wmma(
    const float* __restrict__ X, const float* __restrict__ W,
    const float* __restrict__ bias, float* __restrict__ Y,
    int M, int N, int K)
{
    __shared__ alignas(16) float Af[2][GTM * GTK];   // [m][k], 2 x 16 KiB
    __shared__ alignas(16) float Wf[2][GTK * GTN];   // [k][n], 2 x 8 KiB

    const int tid  = threadIdx.x;
    const int lane = tid & 31;
    const int wave = tid >> 5;
    const int wr   = wave & 3;
    const int wc   = wave >> 2;
    const int m0   = blockIdx.y * GTM;
    const int n0   = blockIdx.x * GTN;
    const int ll   = lane & 15;
    const int lhi  = lane >> 4;

    // Issue async copies for one K-chunk into buffer `buf`:
    // A: 128x32 f32 = 16 KiB -> 4 b128 per thread; W: 32x64 f32 -> 2 per thread.
    auto issue_tile = [&](int buf, int k0) {
        #pragma unroll
        for (int r = 0; r < 4; ++r) {
            int idx = (r * 256 + tid) * 4;         // float index, 16B granules
            int m = idx >> 5, k = idx & 31;
            async_cp16(&X[(size_t)(m0 + m) * K + (k0 + k)], &Af[buf][idx]);
        }
        #pragma unroll
        for (int r = 0; r < 2; ++r) {
            int idx = (r * 256 + tid) * 4;
            int k = idx >> 6, n = idx & 63;
            async_cp16(&W[(size_t)(k0 + k) * N + (n0 + n)], &Wf[buf][idx]);
        }
    };

    v8f acc[2][2];
    #pragma unroll
    for (int tm = 0; tm < 2; ++tm)
        #pragma unroll
        for (int tn = 0; tn < 2; ++tn)
            #pragma unroll
            for (int r = 0; r < 8; ++r) acc[tm][tn][r] = 0.0f;

    issue_tile(0, 0);
    int cur = 0;
    for (int k0 = 0; k0 < K; k0 += GTK) {
        const bool more = (k0 + GTK < K);
        if (more) {
            issue_tile(cur ^ 1, k0 + GTK);
            // 12 outstanding: wait down to 6 => current tile's 6 complete
            asm volatile("s_wait_asynccnt 0x6" ::: "memory");
        } else {
            asm volatile("s_wait_asynccnt 0x0" ::: "memory");
        }
        __syncthreads();

        // A fragments: 16x32 bf16, lane holds row m = ll,
        // DWORD j holds k = {kp, kp+1}, kp = 2j + lhi*16
        BF16x16 afr[2];
        #pragma unroll
        for (int tm = 0; tm < 2; ++tm) {
            const float* p = &Af[cur][(wr * 32 + tm * 16 + ll) * GTK];
            #pragma unroll
            for (int j = 0; j < 8; ++j) {
                int kp = (2 * j) + (lhi << 4);
                afr[tm].w[j] = pk2bf(p[kp], p[kp + 1]);
            }
        }
        // B fragments: 32x16 bf16, lane holds col n = ll,
        // DWORD j holds k = {2j + lhi*16, 2j+1 + lhi*16} (strided rows)
        BF16x16 bfr[2];
        #pragma unroll
        for (int tn = 0; tn < 2; ++tn) {
            int col = wc * 32 + tn * 16 + ll;
            const float* p = &Wf[cur][(lhi << 4) * GTN + col];
            #pragma unroll
            for (int j = 0; j < 8; ++j)
                bfr[tn].w[j] = pk2bf(p[(2 * j) * GTN], p[(2 * j + 1) * GTN]);
        }

        #pragma unroll
        for (int tm = 0; tm < 2; ++tm)
            #pragma unroll
            for (int tn = 0; tn < 2; ++tn)
                acc[tm][tn] = __builtin_amdgcn_wmma_f32_16x16x32_bf16(
                    false, afr[tm].v, false, bfr[tn].v,
                    (short)0, acc[tm][tn], false, false);
        __syncthreads();   // guard LDS buffers before next iteration's issue
        cur ^= 1;
    }

    // C layout: lanes 0-15 vgpr r -> M=r ; lanes 16-31 vgpr r -> M=r+8 ; N = ll
    #pragma unroll
    for (int tm = 0; tm < 2; ++tm)
        #pragma unroll
        for (int tn = 0; tn < 2; ++tn) {
            int col = n0 + wc * 32 + tn * 16 + ll;
            float bvv = bias[col];
            #pragma unroll
            for (int r = 0; r < 8; ++r) {
                int row = m0 + wr * 32 + tm * 16 + r + lhi * 8;
                Y[(size_t)row * N + col] = acc[tm][tn][r] + bvv;
            }
        }
}

// ---------------------------------------------------------------------------
// Mean-pool over sequence: pooled[b,0:D]=mean_s query ; pooled[b,D:2D]=mean_s key
// ---------------------------------------------------------------------------
__global__ __launch_bounds__(256) void pool_kernel(
    const float* __restrict__ q, const float* __restrict__ k,
    float* __restrict__ pooled)
{
    int t = blockIdx.x * blockDim.x + threadIdx.x;
    if (t >= BB * 2 * DD) return;
    int b = t / (2 * DD);
    int c = t % (2 * DD);
    const float* src = (c < DD) ? q : k;
    int d = (c < DD) ? c : c - DD;
    float s = 0.0f;
    for (int ss = 0; ss < SS; ++ss) s += src[((size_t)(b * SS + ss)) * DD + d];
    pooled[t] = s * (1.0f / (float)SS);
}

// gate[b,h] = sigmoid(pooled[b,:] @ Wg[:,h] + bg[h])
__global__ void gate_kernel(const float* __restrict__ pooled,
                            const float* __restrict__ Wg,
                            const float* __restrict__ bg,
                            float* __restrict__ gate)
{
    int t = threadIdx.x;
    if (t >= BB * HH) return;
    int b = t / HH, h = t % HH;
    float s = bg[h];
    for (int d = 0; d < 2 * DD; ++d) s += pooled[b * 2 * DD + d] * Wg[d * HH + h];
    gate[t] = 1.0f / (1.0f + __expf(-s));
}

// ---------------------------------------------------------------------------
// Flash attention (causal) with per-(b,h) gate.
// Grid: B*H*(S/64) blocks; block = 128 threads = 4 waves; wave w owns 16 q-rows.
// K/V tiles (32 keys x 64 dk, f32) staged by async global->LDS copies.
// ---------------------------------------------------------------------------
#define QTILE 64

__global__ __launch_bounds__(128) void flash_attn_wmma(
    const float* __restrict__ Qb, const float* __restrict__ Kb,
    const float* __restrict__ Vb, const float* __restrict__ gate,
    float* __restrict__ ctx)
{
    __shared__ alignas(16) float Kf[32 * 64];   // [key][d] f32, 8 KiB
    __shared__ alignas(16) float Vf[32 * 64];   // [key][d] f32, 8 KiB
    __shared__ float Plds[4][16][32];           // per-wave P bounce, 8 KiB

    const int tid  = threadIdx.x;
    const int lane = tid & 31;
    const int wave = tid >> 5;
    const int ll   = lane & 15;
    const int lhi  = lane >> 4;

    const int nqb = SS / QTILE;                 // 32
    const int qb  = blockIdx.x % nqb;
    const int h   = (blockIdx.x / nqb) % HH;
    const int b   = blockIdx.x / (nqb * HH);
    const int qbase = qb * QTILE;

    // Q A-fragments for this wave's 16 rows, DK=64 -> two 16x32 fragments
    const int qrow = qbase + wave * 16 + ll;
    const float* qptr = Qb + ((size_t)(b * SS + qrow)) * DD + h * DK;
    BF16x16 qa[2];
    #pragma unroll
    for (int c = 0; c < 2; ++c)
        #pragma unroll
        for (int j = 0; j < 8; ++j) {
            int kp = c * 32 + (2 * j) + (lhi << 4);
            qa[c].w[j] = pk2bf(qptr[kp], qptr[kp + 1]);
        }

    v8f out[4];
    #pragma unroll
    for (int t = 0; t < 4; ++t)
        #pragma unroll
        for (int r = 0; r < 8; ++r) out[t][r] = 0.0f;

    float mrow[8], lrow[8];
    #pragma unroll
    for (int r = 0; r < 8; ++r) { mrow[r] = -3.0e38f; lrow[r] = 0.0f; }

    const float scale = 0.125f;        // 1/sqrt(64)
    const int jblocks = (qb + 1) * 2;  // causal: keys up to qbase+63

    for (int j = 0; j < jblocks; ++j) {
        const int jbase = j * 32;
        // Async-stage K and V tiles: 2048 f32 each -> 4 b128 per thread per tile
        #pragma unroll
        for (int r = 0; r < 4; ++r) {
            int idx = (r * 128 + tid) * 4;
            int kk = idx >> 6, d = idx & 63;
            size_t g = ((size_t)(b * SS + jbase + kk)) * DD + h * DK + d;
            async_cp16(&Kb[g], &Kf[idx]);
            async_cp16(&Vb[g], &Vf[idx]);
        }
        asm volatile("s_wait_asynccnt 0x0" ::: "memory");
        __syncthreads();

        // Scores for two 16-key tiles: S = Q(16x64) . K^T(64x16)
        v8f sc[2];
        #pragma unroll
        for (int t = 0; t < 2; ++t) {
            BF16x16 kbf[2];
            #pragma unroll
            for (int c = 0; c < 2; ++c) {
                const float* p = &Kf[(t * 16 + ll) * 64 + c * 32 + (lhi << 4)];
                #pragma unroll
                for (int jj = 0; jj < 8; ++jj)
                    kbf[c].w[jj] = pk2bf(p[2 * jj], p[2 * jj + 1]);
            }
            v8f s;
            #pragma unroll
            for (int r = 0; r < 8; ++r) s[r] = 0.0f;
            s = __builtin_amdgcn_wmma_f32_16x16x32_bf16(false, qa[0].v, false, kbf[0].v,
                                                        (short)0, s, false, false);
            s = __builtin_amdgcn_wmma_f32_16x16x32_bf16(false, qa[1].v, false, kbf[1].v,
                                                        (short)0, s, false, false);
            sc[t] = s;
        }

        // Scale + causal mask + online softmax (row group = 16 lanes)
        #pragma unroll
        for (int r = 0; r < 8; ++r) {
            int qi = qbase + wave * 16 + r + lhi * 8;
            #pragma unroll
            for (int t = 0; t < 2; ++t) {
                int kj = jbase + t * 16 + ll;
                float v = sc[t][r] * scale;
                sc[t][r] = (kj > qi) ? -3.0e38f : v;
            }
            float mb = fmaxf(sc[0][r], sc[1][r]);
            #pragma unroll
            for (int off = 1; off < 16; off <<= 1)
                mb = fmaxf(mb, __shfl_xor(mb, off, 32));
            float mnew  = fmaxf(mrow[r], mb);
            float alpha = __expf(mrow[r] - mnew);
            mrow[r] = mnew;
            float p0 = __expf(sc[0][r] - mnew);
            float p1 = __expf(sc[1][r] - mnew);
            sc[0][r] = p0; sc[1][r] = p1;
            float rs = p0 + p1;
            #pragma unroll
            for (int off = 1; off < 16; off <<= 1)
                rs += __shfl_xor(rs, off, 32);
            lrow[r] = lrow[r] * alpha + rs;
            #pragma unroll
            for (int t = 0; t < 4; ++t) out[t][r] *= alpha;
        }

        // Bounce P (C layout) through LDS to re-pack as an A fragment (16x32)
        #pragma unroll
        for (int t = 0; t < 2; ++t)
            #pragma unroll
            for (int r = 0; r < 8; ++r)
                Plds[wave][r + 8 * lhi][t * 16 + ll] = sc[t][r];
        BF16x16 pf;   // same-wave LDS RAW: compiler inserts s_wait_dscnt
        {
            const float* p = &Plds[wave][ll][0];
            #pragma unroll
            for (int jj = 0; jj < 8; ++jj) {
                int kp = (2 * jj) + (lhi << 4);
                pf.w[jj] = pk2bf(p[kp], p[kp + 1]);
            }
        }

        // out(16x64) += P(16x32) . V(32x64): 4 WMMAs over dk tiles
        #pragma unroll
        for (int tv = 0; tv < 4; ++tv) {
            const float* p = &Vf[(lhi << 4) * 64 + tv * 16 + ll];
            BF16x16 vf;
            #pragma unroll
            for (int jj = 0; jj < 8; ++jj)
                vf.w[jj] = pk2bf(p[(2 * jj) * 64], p[(2 * jj + 1) * 64]);
            out[tv] = __builtin_amdgcn_wmma_f32_16x16x32_bf16(
                false, pf.v, false, vf.v, (short)0, out[tv], false, false);
        }
        __syncthreads();  // protect K/V LDS before next stage
    }

    // Epilogue: normalize, apply gate, write ctx in (B,S,H*DK) layout
    const float g = gate[b * HH + h];
    #pragma unroll
    for (int tv = 0; tv < 4; ++tv)
        #pragma unroll
        for (int r = 0; r < 8; ++r) {
            int qi = qbase + wave * 16 + r + 8 * lhi;
            int dc = h * DK + tv * 16 + ll;
            ctx[((size_t)(b * SS + qi)) * DD + dc] = out[tv][r] / lrow[r] * g;
        }
}

// ---------------------------------------------------------------------------
// Launch: Q/K/V projections -> pool -> gate -> flash attention -> out proj
// Workspace layout (bytes): Q[16M] K[16M] V[16M] CTX[16M] pooled[16K] gate[128]
// ---------------------------------------------------------------------------
extern "C" void kernel_launch(void* const* d_in, const int* in_sizes, int n_in,
                              void* d_out, int out_size, void* d_ws, size_t ws_size,
                              hipStream_t stream) {
    (void)in_sizes; (void)n_in; (void)out_size; (void)ws_size;

    const float* query = (const float*)d_in[0];
    const float* key_  = (const float*)d_in[1];
    const float* value = (const float*)d_in[2];
    // d_in[3] = causal bool mask; implemented analytically in the kernel
    const float* Wq = (const float*)d_in[4];
    const float* bq = (const float*)d_in[5];
    const float* Wk = (const float*)d_in[6];
    const float* bk = (const float*)d_in[7];
    const float* Wv = (const float*)d_in[8];
    const float* bv = (const float*)d_in[9];
    const float* Wo = (const float*)d_in[10];
    const float* bo = (const float*)d_in[11];
    const float* Wg = (const float*)d_in[12];
    const float* bg = (const float*)d_in[13];
    float* out = (float*)d_out;

    const size_t MAT = (size_t)BB * SS * DD * sizeof(float);  // 16 MiB
    char* ws = (char*)d_ws;
    float* Qb     = (float*)(ws + 0 * MAT);
    float* Kb     = (float*)(ws + 1 * MAT);
    float* Vb     = (float*)(ws + 2 * MAT);
    float* ctx    = (float*)(ws + 3 * MAT);
    float* pooled = (float*)(ws + 4 * MAT);
    float* gateb  = (float*)(ws + 4 * MAT + BB * 2 * DD * sizeof(float));

    const int M = BB * SS;   // 4096
    const int N = DD;        // 1024
    const int K = DD;        // 1024
    dim3 ggrid(N / GTN, M / GTM);   // (16, 32)

    gemm_bias_wmma<<<ggrid, 256, 0, stream>>>(query, Wq, bq, Qb, M, N, K);
    gemm_bias_wmma<<<ggrid, 256, 0, stream>>>(key_,  Wk, bk, Kb, M, N, K);
    gemm_bias_wmma<<<ggrid, 256, 0, stream>>>(value, Wv, bv, Vb, M, N, K);

    pool_kernel<<<(BB * 2 * DD + 255) / 256, 256, 0, stream>>>(query, key_, pooled);
    gate_kernel<<<1, 32, 0, stream>>>(pooled, Wg, bg, gateb);

    flash_attn_wmma<<<BB * HH * (SS / QTILE), 128, 0, stream>>>(Qb, Kb, Vb, gateb, ctx);

    gemm_bias_wmma<<<ggrid, 256, 0, stream>>>(ctx, Wo, bo, out, M, N, K);
}
